// DenseSSMLayer_62302795596574
// MI455X (gfx1250) — matrix operand order repaired and checked by reading, not compiled, not run on hardware
//
#include <hip/hip_runtime.h>
#include <hip/hip_bf16.h>
#include <math.h>
#include <cstdint>

typedef _Float16 f16;
typedef __attribute__((ext_vector_type(16))) _Float16 v16h;
typedef __attribute__((ext_vector_type(8)))  _Float16 h8;
typedef __attribute__((ext_vector_type(8)))  float    v8f;
typedef __attribute__((ext_vector_type(4)))  float    fvec4; // POD 16B chunk

// Problem sizes (fixed by the reference)
constexpr int B_  = 4;
constexpr int T_  = 2048;
constexpr int DM_ = 1024;
constexpr int N_  = 128;
constexpr int NN_ = N_ * N_;       // 16384
constexpr int BT_ = B_ * T_;       // 8192
constexpr float INV_SQRT_N = 0.08838834764831845f; // 1/sqrt(128)

// ---- workspace layout (bytes); every section is 256B-aligned ----
constexpr size_t OFF_U16 = 0;                                   // u f16     [BT,DM]   16 MB
constexpr size_t OFF_WAT = OFF_U16 + (size_t)BT_ * DM_ * 2;     // WA^T f16  [NN,DM]   32 MB
constexpr size_t OFF_WDT = OFF_WAT + (size_t)NN_ * DM_ * 2;     // Wd^T f16  [N,DM]
constexpr size_t OFF_WBT = OFF_WDT + (size_t)N_ * DM_ * 2;      // WB^T f16  [N,DM]
constexpr size_t OFF_CT  = OFF_WBT + (size_t)N_ * DM_ * 2;      // C^T  f16  [DM,N]
constexpr size_t OFF_D   = OFF_CT  + (size_t)DM_ * N_ * 2;      // d    f32  [BT,N]    4 MB
constexpr size_t OFF_BT  = OFF_D   + (size_t)BT_ * N_ * 4;      // Bt   f32  [BT,N]    4 MB
constexpr size_t OFF_A16 = OFF_BT  + (size_t)BT_ * N_ * 4;      // A    f16  [BT,N,N]  256 MB
constexpr size_t OFF_HS  = OFF_A16 + (size_t)BT_ * NN_ * 2;     // hs   f16  [BT,N]    2 MB

// ---- WMMA fragment loaders (layouts per CDNA5 ISA sec 7.12.2) ----
// A operand (16x32 f16): lane L holds row L&15; K = koff+{0..7} and koff+16+{0..7}, koff=(L&16)?8:0
// Works for both global and LDS base pointers (addrspace inferred after inlining:
// global callsites -> global_load_b128, shared callsites -> ds_load_b128).
__device__ __forceinline__ v16h load_fragA(const f16* __restrict__ base, int ld, int k0) {
  int lane = threadIdx.x & 31;
  int row  = lane & 15;
  int koff = (lane & 16) ? 8 : 0;
  const f16* p = base + (size_t)row * ld + k0 + koff;
  union { v16h v; fvec4 q[2]; } u;
  u.q[0] = *(const fvec4*)(p);
  u.q[1] = *(const fvec4*)(p + 16);
  return u.v;
}
// B operand (32x16 f16): lane L holds column L&15; K = koff+{0..7} and koff+16+{0..7}, koff=(L&16)?8:0
__device__ __forceinline__ v16h load_fragB(const f16* __restrict__ base, int ld, int k0) {
  int lane = threadIdx.x & 31;
  int col  = lane & 15;
  int koff = (lane & 16) ? 8 : 0;
  const f16* p = base + (size_t)col * ld + k0 + koff;
  union { v16h v; fvec4 q[2]; } u;
  u.q[0] = *(const fvec4*)(p);
  u.q[1] = *(const fvec4*)(p + 16);
  return u.v;
}

// ---- async global -> LDS block copy (gfx1250 ASYNCcnt path) ----
// Copies 16 bytes per thread per step; INST_OFFSET applies to both LDS and global
// addresses, which matches our layout (both advance together).
__device__ __forceinline__ void async_copy_wait() {
  asm volatile("s_wait_asynccnt 0x0" ::: "memory");
}

// ---- convert / transpose-convert (prep pass) ----
__global__ void k_cvt(f16* __restrict__ dst, const float* __restrict__ src, size_t n) {
  size_t i = (size_t)blockIdx.x * blockDim.x + threadIdx.x;
  if (i < n) dst[i] = (f16)src[i];
}
// src [R,C] row-major f32 -> dst [C,R] row-major f16
__global__ void k_cvtT(f16* __restrict__ dst, const float* __restrict__ src, int R, int C) {
  size_t i = (size_t)blockIdx.x * blockDim.x + threadIdx.x;
  size_t total = (size_t)R * C;
  if (i < total) {
    int r = (int)(i % (size_t)R);
    int c = (int)(i / (size_t)R);
    dst[i] = (f16)src[(size_t)r * C + c];
  }
}

// ---- d = sigmoid(u@Wd + bd), Bt = u@WB + bB  (16 waves: 8 for d, 8 for Bt) ----
__global__ void __launch_bounds__(512) k_dB(
    const f16* __restrict__ u16, const f16* __restrict__ Wdt,
    const f16* __restrict__ WBt, const float* __restrict__ bd,
    const float* __restrict__ bB, float* __restrict__ dbuf,
    float* __restrict__ Btbuf) {
  __shared__ f16 tileA[16 * DM_];                 // 32 KB: u rows m0..m0+15
  int tid  = threadIdx.x;
  int wave = tid >> 5;
  int m0   = blockIdx.x * 16;
  // async stage: 512 threads x 16B x 4 steps = 32 KB
  {
    unsigned    ldsb = (unsigned)(uintptr_t)(void*)tileA + (unsigned)tid * 16u;
    const char* gb   = (const char*)(u16 + (size_t)m0 * DM_) + (size_t)tid * 16;
#pragma unroll
    for (int it = 0; it < 4; ++it)
      asm volatile("global_load_async_to_lds_b128 %0, %1, off offset:%2"
                   :: "v"(ldsb), "v"(gb), "i"(it * 8192) : "memory");
    async_copy_wait();
    __syncthreads();
  }
  bool isD = wave < 8;
  int n0   = (isD ? wave : wave - 8) * 16;
  const f16* W = isD ? Wdt : WBt;
  v8f acc = {};
  for (int kk = 0; kk < DM_; kk += 32) {
    v16h a = load_fragA(tileA, DM_, kk);          // ds_load_b128 path
    v16h b = load_fragB(W + (size_t)n0 * DM_, DM_, kk);
    acc = __builtin_amdgcn_wmma_f32_16x16x32_f16(false, a, false, b, (short)0, acc, false, false);
  }
  int lane = tid & 31;
  int rowadd = (lane & 16) ? 8 : 0;
  int colc   = lane & 15;
#pragma unroll
  for (int r = 0; r < 8; ++r) {
    int gm = m0 + r + rowadd;
    int gc = n0 + colc;
    float v = acc[r];
    if (isD) {
      dbuf [(size_t)gm * N_ + gc] = 1.0f / (1.0f + __expf(-(v + bd[gc])));
    } else {
      Btbuf[(size_t)gm * N_ + gc] = v + bB[gc];
    }
  }
}

// ---- A = tanh(u@WA + bA)/sqrt(N), diag <- d ; stored f16 [BT,N,N] ----
// wave: 16x64 tile; block: 8 waves -> 16x512; grid (NN/512, BT/16)
__global__ void __launch_bounds__(256) k_buildA(
    const f16* __restrict__ u16, const f16* __restrict__ WAt,
    const float* __restrict__ bA, const float* __restrict__ dbuf,
    f16* __restrict__ A16) {
  __shared__ f16 tileA[16 * DM_];                 // 32 KB: u rows m0..m0+15
  int tid   = threadIdx.x;
  int wave  = tid >> 5;
  int m0    = blockIdx.y * 16;
  int nbase = blockIdx.x * 512 + wave * 64;
  // async stage: 256 threads x 16B x 8 steps = 32 KB
  {
    unsigned    ldsb = (unsigned)(uintptr_t)(void*)tileA + (unsigned)tid * 16u;
    const char* gb   = (const char*)(u16 + (size_t)m0 * DM_) + (size_t)tid * 16;
#pragma unroll
    for (int it = 0; it < 8; ++it)
      asm volatile("global_load_async_to_lds_b128 %0, %1, off offset:%2"
                   :: "v"(ldsb), "v"(gb), "i"(it * 4096) : "memory");
    async_copy_wait();
    __syncthreads();
  }
  v8f acc[4] = {};
  for (int kk = 0; kk < DM_; kk += 32) {
    v16h a = load_fragA(tileA, DM_, kk);          // ds_load_b128 path
#pragma unroll
    for (int j = 0; j < 4; ++j) {
      v16h b = load_fragB(WAt + (size_t)(nbase + 16 * j) * DM_, DM_, kk);
      acc[j] = __builtin_amdgcn_wmma_f32_16x16x32_f16(false, a, false, b, (short)0, acc[j], false, false);
    }
  }
  int lane = tid & 31;
  int rowadd = (lane & 16) ? 8 : 0;
  int colc   = lane & 15;
#pragma unroll
  for (int j = 0; j < 4; ++j) {
#pragma unroll
    for (int r = 0; r < 8; ++r) {
      int gm = m0 + r + rowadd;
      int gc = nbase + 16 * j + colc;
      float v = tanhf(acc[j][r] + bA[gc]) * INV_SQRT_N;
      if (gc % (N_ + 1) == 0)                 // diagonal element (i==j), i = gc/129
        v = dbuf[(size_t)gm * N_ + gc / (N_ + 1)];
      A16[(size_t)gm * NN_ + gc] = (f16)v;
    }
  }
}

// ---- sequential scan: h_t = A_t h_{t-1} + B_t ; one block per batch ----
__global__ void k_scan(const f16* __restrict__ A16, const float* __restrict__ Btbuf,
                       f16* __restrict__ hs16) {
  __shared__ float h[N_];
  int b = blockIdx.x, i = threadIdx.x;
  h[i] = 0.0f;
  __syncthreads();
  for (int t = 0; t < T_; ++t) {
    size_t bt = (size_t)b * T_ + t;
    const f16* Arow = A16 + bt * NN_ + (size_t)i * N_;
    if (t + 1 < T_) __builtin_prefetch(Arow + NN_, 0, 1);   // global_prefetch next A_t row
    float sum = 0.0f;
#pragma unroll
    for (int c = 0; c < 16; ++c) {
      h8 av = *(const h8*)(Arow + c * 8);
#pragma unroll
      for (int e = 0; e < 8; ++e) sum += (float)av[e] * h[c * 8 + e];
    }
    sum += Btbuf[bt * N_ + i];
    __syncthreads();
    h[i] = sum;
    hs16[bt * N_ + i] = (f16)sum;
    __syncthreads();
  }
}

// ---- y = hs @ C + D_skip * u ; wave 16x64 tile, block 16x512, grid (DM/512, BT/16) ----
__global__ void __launch_bounds__(256) k_out(
    const f16* __restrict__ hs16, const f16* __restrict__ Ct,
    const float* __restrict__ Dskip, const float* __restrict__ ufp,
    float* __restrict__ y) {
  __shared__ f16 tileA[16 * N_];                  // 4 KB: hs rows m0..m0+15
  int tid   = threadIdx.x;
  int wave  = tid >> 5;
  int m0    = blockIdx.y * 16;
  int nbase = blockIdx.x * 512 + wave * 64;
  // async stage: 256 threads x 16B = 4 KB
  {
    unsigned    ldsb = (unsigned)(uintptr_t)(void*)tileA + (unsigned)tid * 16u;
    const char* gb   = (const char*)(hs16 + (size_t)m0 * N_) + (size_t)tid * 16;
    asm volatile("global_load_async_to_lds_b128 %0, %1, off"
                 :: "v"(ldsb), "v"(gb) : "memory");
    async_copy_wait();
    __syncthreads();
  }
  v8f acc[4] = {};
  for (int kk = 0; kk < N_; kk += 32) {
    v16h a = load_fragA(tileA, N_, kk);           // ds_load_b128 path
#pragma unroll
    for (int j = 0; j < 4; ++j) {
      v16h b = load_fragB(Ct + (size_t)(nbase + 16 * j) * N_, N_, kk);
      acc[j] = __builtin_amdgcn_wmma_f32_16x16x32_f16(false, a, false, b, (short)0, acc[j], false, false);
    }
  }
  int lane = tid & 31;
  int rowadd = (lane & 16) ? 8 : 0;
  int colc   = lane & 15;
#pragma unroll
  for (int j = 0; j < 4; ++j) {
#pragma unroll
    for (int r = 0; r < 8; ++r) {
      int gm = m0 + r + rowadd;
      int gc = nbase + 16 * j + colc;
      y[(size_t)gm * DM_ + gc] = acc[j][r] + Dskip[gc] * ufp[(size_t)gm * DM_ + gc];
    }
  }
}

extern "C" void kernel_launch(void* const* d_in, const int* in_sizes, int n_in,
                              void* d_out, int out_size, void* d_ws, size_t ws_size,
                              hipStream_t stream) {
  const float* u  = (const float*)d_in[0];
  const float* Wd = (const float*)d_in[1];
  const float* bd = (const float*)d_in[2];
  const float* WA = (const float*)d_in[3];
  const float* bA = (const float*)d_in[4];
  const float* WB = (const float*)d_in[5];
  const float* bB = (const float*)d_in[6];
  const float* C  = (const float*)d_in[7];
  const float* Dk = (const float*)d_in[8];

  char* ws = (char*)d_ws;
  f16*   u16   = (f16*)  (ws + OFF_U16);
  f16*   WAt   = (f16*)  (ws + OFF_WAT);
  f16*   Wdt   = (f16*)  (ws + OFF_WDT);
  f16*   WBt   = (f16*)  (ws + OFF_WBT);
  f16*   Ct    = (f16*)  (ws + OFF_CT);
  float* dbuf  = (float*)(ws + OFF_D);
  float* Btbuf = (float*)(ws + OFF_BT);
  f16*   A16   = (f16*)  (ws + OFF_A16);
  f16*   hs16  = (f16*)  (ws + OFF_HS);

  // 1) convert / transpose weights+activations to f16 WMMA-friendly layouts
  {
    size_t n = (size_t)BT_ * DM_;
    k_cvt<<<(unsigned)((n + 255) / 256), 256, 0, stream>>>(u16, u, n);
  }
  {
    size_t n = (size_t)DM_ * NN_;
    k_cvtT<<<(unsigned)((n + 255) / 256), 256, 0, stream>>>(WAt, WA, DM_, NN_);
  }
  {
    size_t n = (size_t)DM_ * N_;
    k_cvtT<<<(unsigned)((n + 255) / 256), 256, 0, stream>>>(Wdt, Wd, DM_, N_);
    k_cvtT<<<(unsigned)((n + 255) / 256), 256, 0, stream>>>(WBt, WB, DM_, N_);
  }
  {
    size_t n = (size_t)N_ * DM_;
    k_cvtT<<<(unsigned)((n + 255) / 256), 256, 0, stream>>>(Ct, C, N_, DM_);
  }

  // 2) gates d and input projection Bt (WMMA, async-LDS staged A operand)
  k_dB<<<BT_ / 16, 512, 0, stream>>>(u16, Wdt, WBt, bd, bB, dbuf, Btbuf);

  // 3) dominant GEMM: build transition matrices A (WMMA + tanh epilogue + diag override)
  k_buildA<<<dim3(NN_ / 512, BT_ / 16), 256, 0, stream>>>(u16, WAt, bA, dbuf, A16);

  // 4) sequential recurrence (latency path), one WGP per batch
  k_scan<<<B_, N_, 0, stream>>>(A16, Btbuf, hs16);

  // 5) output projection + skip (WMMA, async-LDS staged A operand)
  k_out<<<dim3(DM_ / 512, BT_ / 16), 256, 0, stream>>>(hs16, Ct, Dk, u, (float*)d_out);
}
